// QNetwork_32615981645978
// MI455X (gfx1250) — matrix operand order repaired and checked by reading, not compiled, not run on hardware
//
#include <hip/hip_runtime.h>
#include <math.h>

// ---------------- problem dimensions (compile-time) ----------------
#define B_      512
#define S_      6
#define S2_     32
#define H_      256
#define NH_     8
#define HD_     32
#define DF_     2048
#define NL_     3
#define NLAYER_ 3
#define MROWS   (B_*S_)    // 3072 transformer token rows
#define MACT    (B_*S2_)   // 16384 action rows

typedef _Float16 f16;
typedef __attribute__((ext_vector_type(16))) _Float16 v16h;
typedef __attribute__((ext_vector_type(4)))  _Float16 h4;
typedef __attribute__((ext_vector_type(8)))  float    v8f;

union Frag { unsigned int u[8]; v16h v; };

// =====================================================================
// Generic WMMA GEMM:  C[M,N] = A[M,K] x op(B) + bias, optional ReLU.
//   opB==0: B is [N,K] row-major with row stride ldb  (C = A * B^T)
//   opB==1: B is [K,N] row-major with row stride ldb  (C = A * B)
// A is f32 (lda=K), C is f32 (ldc=N). f32 -> f16 conversion on stage,
// f32 accumulate via v_wmma_f32_16x16x32_f16.
// Block: 256 threads = 8 waves; block tile 128x128; wave tile 32x64.
// Software-pipelined: K-tile t+1 prefetched into VGPRs (b128 loads)
// while WMMAs consume K-tile t from LDS.
// M,N multiples of 128; K multiple of 32 (true for every call here).
// =====================================================================
__global__ __launch_bounds__(256) void gemm_wmma(
    const float* __restrict__ A, const float* __restrict__ B,
    const float* __restrict__ bias, float* __restrict__ C,
    int M, int N, int K, int ldb, int opB, int relu)
{
  __shared__ _Float16 As[128][36];   // 72B rows: 8B-aligned h4 stores, conflict-free frag reads
  __shared__ _Float16 Bs[32][132];   // Bs[k][n]; 264B rows

  const int tid  = threadIdx.x;
  const int lane = tid & 31;
  const int wid  = tid >> 5;
  const int wm   = wid & 3;          // 4 wave-rows
  const int wn   = wid >> 2;         // 2 wave-cols
  const int m0   = blockIdx.y * 128;
  const int n0   = blockIdx.x * 128;
  const int mwave = wm * 32;
  const int nwave = wn * 64;
  const int lrow  = lane & 15;
  const int hi    = (lane >> 4) << 3;   // 0 or 8

  float4 a4[4], b4[4];               // prefetch registers (one K-tile)

  auto loadAB = [&](int k0) {
#pragma unroll
    for (int i = 0; i < 4; ++i) {          // A tile 128x32 as 1024 float4
      int idx = tid + i * 256;
      int r = idx >> 3, c4 = idx & 7;
      a4[i] = *(const float4*)(A + (size_t)(m0 + r) * K + k0 + c4 * 4);
    }
    if (opB == 0) {
#pragma unroll
      for (int i = 0; i < 4; ++i) {        // B tile 128n x 32k, vector along k
        int idx = tid + i * 256;
        int n = idx >> 3, c4 = idx & 7;
        b4[i] = *(const float4*)(B + (size_t)(n0 + n) * ldb + k0 + c4 * 4);
      }
    } else {
#pragma unroll
      for (int i = 0; i < 4; ++i) {        // B tile 32k x 128n, vector along n
        int idx = tid + i * 256;
        int kk = idx >> 5, n4 = idx & 31;
        b4[i] = *(const float4*)(B + (size_t)(k0 + kk) * ldb + n0 + n4 * 4);
      }
    }
  };

  auto storeAB = [&]() {
#pragma unroll
    for (int i = 0; i < 4; ++i) {
      int idx = tid + i * 256;
      int r = idx >> 3, c4 = idx & 7;
      h4 t;
      t.x = (_Float16)a4[i].x; t.y = (_Float16)a4[i].y;
      t.z = (_Float16)a4[i].z; t.w = (_Float16)a4[i].w;
      *(h4*)(&As[r][c4 * 4]) = t;          // ds_store_b64
    }
    if (opB == 0) {                        // transpose into Bs[k][n]
#pragma unroll
      for (int i = 0; i < 4; ++i) {
        int idx = tid + i * 256;
        int n = idx >> 3, c4 = idx & 7;
        Bs[c4 * 4 + 0][n] = (_Float16)b4[i].x;
        Bs[c4 * 4 + 1][n] = (_Float16)b4[i].y;
        Bs[c4 * 4 + 2][n] = (_Float16)b4[i].z;
        Bs[c4 * 4 + 3][n] = (_Float16)b4[i].w;
      }
    } else {
#pragma unroll
      for (int i = 0; i < 4; ++i) {
        int idx = tid + i * 256;
        int kk = idx >> 5, n4 = idx & 31;
        h4 t;
        t.x = (_Float16)b4[i].x; t.y = (_Float16)b4[i].y;
        t.z = (_Float16)b4[i].z; t.w = (_Float16)b4[i].w;
        *(h4*)(&Bs[kk][n4 * 4]) = t;       // ds_store_b64
      }
    }
  };

  v8f zero = {0.f,0.f,0.f,0.f,0.f,0.f,0.f,0.f};
  v8f acc[2][4];
#pragma unroll
  for (int i = 0; i < 2; ++i)
#pragma unroll
    for (int j = 0; j < 4; ++j) acc[i][j] = zero;

  loadAB(0);
  storeAB();

  for (int k0 = 0; k0 < K; k0 += 32) {
    __syncthreads();                       // LDS tile ready
    const bool more = (k0 + 32) < K;
    if (more) loadAB(k0 + 32);             // prefetch next tile (overlaps WMMA)

    // A fragments (16x32, ISA 16-bit A layout: lane=row, VGPR j holds K pair
    // {2j | 16+2(j-4)} + (lane>=16 ? 8 : 0))
    Frag fa[2];
#pragma unroll
    for (int sm = 0; sm < 2; ++sm) {
      const _Float16* arow = &As[mwave + sm * 16 + lrow][0];
#pragma unroll
      for (int j = 0; j < 8; ++j) {
        int kb = ((j < 4) ? (2 * j) : (8 + 2 * j)) + hi;
        fa[sm].u[j] = *(const unsigned int*)(arow + kb);
      }
    }
    // B fragments (32x16: lane holds column N=lane%16; VGPR j holds same K pair as A)
#pragma unroll
    for (int sn = 0; sn < 4; ++sn) {
      Frag fb;
      const _Float16* bcol = &Bs[0][nwave + sn * 16 + lrow];
#pragma unroll
      for (int j = 0; j < 8; ++j) {
        int kb = ((j < 4) ? (2 * j) : (8 + 2 * j)) + hi;
        fb.v[2 * j] = bcol[kb * 132];
        fb.v[2 * j + 1] = bcol[(kb + 1) * 132];
      }
#pragma unroll
      for (int sm = 0; sm < 2; ++sm)
        acc[sm][sn] = __builtin_amdgcn_wmma_f32_16x16x32_f16(
            false, fa[sm].v, false, fb.v, (short)0, acc[sm][sn], false, false);
    }
    __syncthreads();                       // compute done, LDS reusable
    if (more) storeAB();
  }

  // store: C/D layout -> lane(0-15): N=lane, M=r; lane(16-31): N=lane-16, M=8+r
#pragma unroll
  for (int sm = 0; sm < 2; ++sm) {
#pragma unroll
    for (int sn = 0; sn < 4; ++sn) {
      int col = n0 + nwave + sn * 16 + (lane & 15);
      int rowb = m0 + mwave + sm * 16 + hi;
      float bv = bias ? bias[col] : 0.f;
#pragma unroll
      for (int r = 0; r < 8; ++r) {
        float v = acc[sm][sn][r] + bv;
        if (relu) v = fmaxf(v, 0.f);
        C[(size_t)(rowb + r) * N + col] = v;
      }
    }
  }
}

// =====================================================================
// Positional encoding: x[l,n,c] = states + PE(n,c)
// =====================================================================
__global__ void posenc_kernel(const float* __restrict__ states, float* __restrict__ x)
{
  int idx = blockIdx.x * 256 + threadIdx.x;    // 3072*256 total
  int c = idx & 255;
  int row = idx >> 8;
  int n = row % S_;
  float j2 = (float)(c & ~1);
  float ang = (float)n * expf(-j2 * (9.2103403719761836f / 256.f)); // ln(1e4)
  float pe = (c & 1) ? cosf(ang) : sinf(ang);
  x[idx] = states[idx] + pe;
}

// =====================================================================
// Batch-axis attention for one (n,h), 16 query rows per block. f32.
// scores = (Q*scale) K^T ; softmax over 512 keys ; ctx = P V
// =====================================================================
__global__ __launch_bounds__(256) void attn_kernel(
    const float* __restrict__ qkv, float* __restrict__ ctx)
{
  const int nh = blockIdx.x;           // 0..47
  const int n = nh >> 3, h = nh & 7;
  const int l0 = blockIdx.y * 16;
  const int tid = threadIdx.x;
  const int wave = tid >> 5, lane = tid & 31;
  const float scale = 0.17677669529663687f;   // 1/sqrt(32)

  __shared__ float qs[16][33];
  __shared__ float kv[64][33];
  __shared__ float sc[16][513];
  __shared__ float dinv[16];

#pragma unroll
  for (int i = 0; i < 2; ++i) {
    int idx = tid + i * 256;
    int r = idx >> 5, d = idx & 31;
    qs[r][d] = qkv[((l0 + r) * S_ + n) * 768 + h * 32 + d] * scale;
  }
  // scores
  for (int mt = 0; mt < 8; ++mt) {
    __syncthreads();
#pragma unroll
    for (int i = 0; i < 8; ++i) {
      int idx = tid + i * 256;
      int mm = idx >> 5, d = idx & 31;
      kv[mm][d] = qkv[((mt * 64 + mm) * S_ + n) * 768 + 256 + h * 32 + d];
    }
    __syncthreads();
#pragma unroll
    for (int i = 0; i < 4; ++i) {
      int idx = tid + i * 256;
      int r = idx >> 6, mm = idx & 63;
      float s = 0.f;
#pragma unroll
      for (int d = 0; d < 32; ++d) s += qs[r][d] * kv[mm][d];
      sc[r][mt * 64 + mm] = s;
    }
  }
  __syncthreads();
  // softmax: 8 waves x 2 rows
#pragma unroll
  for (int rr = 0; rr < 2; ++rr) {
    int r = wave * 2 + rr;
    float mx = -1e30f;
#pragma unroll
    for (int j = 0; j < 16; ++j) mx = fmaxf(mx, sc[r][lane + j * 32]);
    for (int off = 16; off; off >>= 1) mx = fmaxf(mx, __shfl_xor(mx, off, 32));
    float sum = 0.f;
#pragma unroll
    for (int j = 0; j < 16; ++j) {
      float e = __expf(sc[r][lane + j * 32] - mx);
      sc[r][lane + j * 32] = e;
      sum += e;
    }
    for (int off = 16; off; off >>= 1) sum += __shfl_xor(sum, off, 32);
    if (lane == 0) dinv[r] = 1.0f / sum;
  }
  // ctx = P V
  float acc[2] = {0.f, 0.f};
  for (int mt = 0; mt < 8; ++mt) {
    __syncthreads();
#pragma unroll
    for (int i = 0; i < 8; ++i) {
      int idx = tid + i * 256;
      int mm = idx >> 5, d = idx & 31;
      kv[mm][d] = qkv[((mt * 64 + mm) * S_ + n) * 768 + 512 + h * 32 + d];
    }
    __syncthreads();
#pragma unroll
    for (int j = 0; j < 2; ++j) {
      int oi = tid + j * 256;
      int r = oi >> 5, d = oi & 31;
      float a = 0.f;
#pragma unroll
      for (int mm = 0; mm < 64; ++mm) a += sc[r][mt * 64 + mm] * kv[mm][d];
      acc[j] += a;
    }
  }
  __syncthreads();
#pragma unroll
  for (int j = 0; j < 2; ++j) {
    int oi = tid + j * 256;
    int r = oi >> 5, d = oi & 31;
    ctx[((l0 + r) * S_ + n) * H_ + h * 32 + d] = acc[j] * dinv[r];
  }
}

// ---------------- block reduce (blockDim == 256) ----------------
__device__ inline float block_reduce_sum256(float v, float* sb)
{
  int t = threadIdx.x;
  sb[t] = v;
  __syncthreads();
  for (int s = 128; s > 0; s >>= 1) {
    if (t < s) sb[t] += sb[t + s];
    __syncthreads();
  }
  float r = sb[0];
  __syncthreads();
  return r;
}

// =====================================================================
// out = LayerNorm(x + y) * g + b   (row-wise, H=256, one block per row)
// =====================================================================
__global__ void add_ln_kernel(const float* __restrict__ x, const float* __restrict__ y,
                              const float* __restrict__ g, const float* __restrict__ b,
                              float* __restrict__ out)
{
  __shared__ float sb[256];
  int row = blockIdx.x, t = threadIdx.x;
  float v = x[row * H_ + t] + y[row * H_ + t];
  float mu = block_reduce_sum256(v, sb) * (1.f / H_);
  float d = v - mu;
  float var = block_reduce_sum256(d * d, sb) * (1.f / H_);
  float rstd = rsqrtf(var + 1e-5f);
  out[row * H_ + t] = d * rstd * g[t] + b[t];
}

// =====================================================================
// Factored attention-aggregate. One block per (b,q):
//   a[k]  = ReLU( Wa2 . ReLU(P[b,q]+Q[b,k]) + ba2 )   (ba1 folded into P)
//   e[k]  = mask ? exp(a) : 0 ;  w = e / clip(sum e, 2e-15)
//   SF[b,q,:] = sum_k w[k] * out[b,k,:]
// =====================================================================
__global__ void aggregate_kernel(
    const float* __restrict__ P, const float* __restrict__ Q,
    const float* __restrict__ Wa2, const float* __restrict__ ba2,
    const float* __restrict__ amask, const float* __restrict__ smask,
    const float* __restrict__ xout, float* __restrict__ SF)
{
  __shared__ float sb[256];
  __shared__ float ek[S_];
  int bq = blockIdx.x;          // b*32 + q
  int b = bq >> 5;
  int t = threadIdx.x;
  float p = P[bq * H_ + t];
  float w2 = Wa2[t];
  for (int k = 0; k < S_; ++k) {
    float hv = fmaxf(p + Q[(b * S_ + k) * H_ + t], 0.f);
    float s = block_reduce_sum256(w2 * hv, sb);
    if (t == 0) {
      float a = fmaxf(s + ba2[0], 0.f);
      float mk = amask[bq] * smask[b * S_ + k];
      ek[k] = (mk > 0.f) ? expf(a) : 0.f;
    }
    __syncthreads();
  }
  float tot = ek[0] + ek[1] + ek[2] + ek[3] + ek[4] + ek[5];
  float inv = 1.f / fmaxf(tot, 2e-15f);
  float sf = 0.f;
#pragma unroll
  for (int k = 0; k < S_; ++k) sf += ek[k] * inv * xout[(b * S_ + k) * H_ + t];
  SF[bq * H_ + t] = sf;
}

// =====================================================================
// adv[b,s,l] = dot(U[b,s,:], SF[b,s,:]) + blab[l]   (one wave per (b,s))
// =====================================================================
__global__ void advdot_kernel(const float* __restrict__ U, const float* __restrict__ SF,
                              const float* __restrict__ blab, float* __restrict__ advb, int l)
{
  int gt = blockIdx.x * 256 + threadIdx.x;
  int bs = gt >> 5;             // 0..16383
  int lane = threadIdx.x & 31;
  float s = 0.f;
#pragma unroll
  for (int j = 0; j < 8; ++j) {
    int i = lane + j * 32;
    s += U[bs * H_ + i] * SF[bs * H_ + i];
  }
  for (int off = 16; off; off >>= 1) s += __shfl_xor(s, off, 32);
  if (lane == 0) advb[bs * NL_ + l] = s + blab[l];
}

// =====================================================================
// Dueling head: q = val + adv - mean(adv).  One block (128 thr) per b.
// =====================================================================
__global__ void final_kernel(const float* __restrict__ SF, const float* __restrict__ amask,
                             const float* __restrict__ Wv, const float* __restrict__ bv,
                             const float* __restrict__ advb, float* __restrict__ out)
{
  __shared__ float sb[128];
  __shared__ float advl[S2_ * NL_];
  __shared__ float scal[2];
  int b = blockIdx.x, t = threadIdx.x;
  if (t == 0) {
    float an = 0.f;
    for (int s = 0; s < S2_; ++s) an += amask[b * S2_ + s];
    scal[0] = an;
  }
  __syncthreads();
  float anum = scal[0];
  float part = 0.f;
#pragma unroll
  for (int hh = 0; hh < 2; ++hh) {
    int h = t + hh * 128;
    float m = 0.f;
    for (int q = 0; q < S2_; ++q) m += SF[(b * S2_ + q) * H_ + h];
    part += (m / anum) * Wv[h];
  }
  sb[t] = part;
  __syncthreads();
  for (int s = 64; s > 0; s >>= 1) {
    if (t < s) sb[t] += sb[t + s];
    __syncthreads();
  }
  float val = sb[0] + bv[0];
  if (t < S2_ * NL_) advl[t] = advb[b * (S2_ * NL_) + t];
  __syncthreads();
  if (t == 0) {
    float s = 0.f;
    for (int i = 0; i < S2_ * NL_; ++i) s += advl[i];
    scal[1] = s * (1.f / (S2_ * NL_));
  }
  __syncthreads();
  float advmean = scal[1];
  if (t < S2_ * NL_) out[b * (S2_ * NL_) + t] = val + advl[t] - advmean;
}

// =====================================================================
extern "C" void kernel_launch(void* const* d_in, const int* in_sizes, int n_in,
                              void* d_out, int out_size, void* d_ws, size_t ws_size,
                              hipStream_t stream)
{
  (void)in_sizes; (void)n_in; (void)out_size; (void)ws_size;
  const float* states = (const float*)d_in[0];
  const float* smask  = (const float*)d_in[1];
  const float* actions= (const float*)d_in[2];
  const float* amask  = (const float*)d_in[3];
  const float* Wqkv   = (const float*)d_in[4];
  const float* bqkv   = (const float*)d_in[5];
  const float* Wo     = (const float*)d_in[6];
  const float* bo     = (const float*)d_in[7];
  const float* ln1g   = (const float*)d_in[8];
  const float* ln1b   = (const float*)d_in[9];
  const float* W1     = (const float*)d_in[10];
  const float* b1     = (const float*)d_in[11];
  const float* W2     = (const float*)d_in[12];
  const float* b2     = (const float*)d_in[13];
  const float* ln2g   = (const float*)d_in[14];
  const float* ln2b   = (const float*)d_in[15];
  const float* Wa1    = (const float*)d_in[16];
  const float* ba1    = (const float*)d_in[17];
  const float* Wa2    = (const float*)d_in[18];
  const float* ba2    = (const float*)d_in[19];
  const float* Wv     = (const float*)d_in[20];
  const float* bv     = (const float*)d_in[21];
  const float* Wlab   = (const float*)d_in[22];
  const float* blab   = (const float*)d_in[23];

  size_t off = 0;
  char* wsb = (char*)d_ws;
  auto alloc = [&](size_t bytes) -> float* {
    float* p = (float*)(wsb + off);
    off += (bytes + 255) & ~(size_t)255;
    return p;
  };
  float* x    = alloc((size_t)MROWS * H_ * 4);       // token activations
  float* qkvb = alloc((size_t)MROWS * 3 * H_ * 4);   // qkv
  float* ctx  = alloc((size_t)MROWS * H_ * 4);
  float* t1   = alloc((size_t)MROWS * H_ * 4);       // sa / ff output
  float* hmid = alloc((size_t)MROWS * DF_ * 4);      // FFN hidden
  float* P    = alloc((size_t)MACT * H_ * 4);        // actions * Wa1L^T + ba1
  float* Qagg = alloc((size_t)MROWS * H_ * 4);       // out * Wa1R^T
  float* SF   = alloc((size_t)MACT * H_ * 4);        // states_feat
  float* U    = alloc((size_t)MACT * H_ * 4);        // actions * Wlab[l]
  float* advb = alloc((size_t)B_ * S2_ * NL_ * 4);   // adv

  // x = states + posenc
  posenc_kernel<<<MROWS, 256, 0, stream>>>(states, x);

  for (int i = 0; i < NLAYER_; ++i) {
    // QKV: [3072,768] = x * Wqkv^T + bqkv
    gemm_wmma<<<dim3(6, 24), 256, 0, stream>>>(
        x, Wqkv + (size_t)i * 3 * H_ * H_, bqkv + (size_t)i * 3 * H_, qkvb,
        MROWS, 3 * H_, H_, H_, 0, 0);
    // batch-axis attention
    attn_kernel<<<dim3(48, 32), 256, 0, stream>>>(qkvb, ctx);
    // output projection
    gemm_wmma<<<dim3(2, 24), 256, 0, stream>>>(
        ctx, Wo + (size_t)i * H_ * H_, bo + (size_t)i * H_, t1,
        MROWS, H_, H_, H_, 0, 0);
    add_ln_kernel<<<MROWS, 256, 0, stream>>>(x, t1, ln1g + i * H_, ln1b + i * H_, x);
    // FFN
    gemm_wmma<<<dim3(16, 24), 256, 0, stream>>>(
        x, W1 + (size_t)i * DF_ * H_, b1 + (size_t)i * DF_, hmid,
        MROWS, DF_, H_, H_, 0, 1);
    gemm_wmma<<<dim3(2, 24), 256, 0, stream>>>(
        hmid, W2 + (size_t)i * H_ * DF_, b2 + (size_t)i * H_, t1,
        MROWS, H_, DF_, DF_, 0, 0);
    add_ln_kernel<<<MROWS, 256, 0, stream>>>(x, t1, ln2g + i * H_, ln2b + i * H_, x);
  }

  // factored aggregate MLP: P = actions * Wa1[:, :256]^T + ba1 ; Q = out * Wa1[:, 256:]^T
  gemm_wmma<<<dim3(2, 128), 256, 0, stream>>>(actions, Wa1, ba1, P,
                                              MACT, H_, H_, 2 * H_, 0, 0);
  gemm_wmma<<<dim3(2, 24), 256, 0, stream>>>(x, Wa1 + H_, nullptr, Qagg,
                                             MROWS, H_, H_, 2 * H_, 0, 0);
  aggregate_kernel<<<MACT, 256, 0, stream>>>(P, Qagg, Wa2, ba2, amask, smask, x, SF);

  // bilinear advantage per label: U = actions * Wlab[l]  (opB=1), then row dots
  for (int l = 0; l < NL_; ++l) {
    gemm_wmma<<<dim3(2, 128), 256, 0, stream>>>(
        actions, Wlab + (size_t)l * H_ * H_, nullptr, U,
        MACT, H_, H_, H_, 1, 0);
    advdot_kernel<<<MACT / 8, 256, 0, stream>>>(U, SF, blab, advb, l);
  }

  // dueling combine
  final_kernel<<<B_, 128, 0, stream>>>(SF, amask, Wv, bv, advb, (float*)d_out);
}
